// MonarchAttention_2920577761658
// MI455X (gfx1250) — compile-verified, matrix operand order checked by reading, not run on hardware
//
#include <hip/hip_runtime.h>

// MonarchAttention on MI455X (gfx1250). All phases are batched 64x64x64 GEMMs
// on v_wmma_f32_16x16x32_bf16 (f32 accum). Intermediates stored bf16 (problem
// is HBM-bound: ~270MB mandatory traffic vs ~30 GFLOP). All WMMA operands are
// staged in LDS K-contiguous (A:[m][k], B:[n][k]) so every fragment is built
// from two 16B ds_load_b128 per lane instead of 16 scalar ds_load_u16.

#define TILE_LD 72   // bf16 LDS row stride: 144B rows -> 16B-chunk stride 9 mod 16,
                     // conflict-free b128 loads across 16 lanes; 16B aligned.
#define S_LD    65   // f32 score LDS row stride

typedef __attribute__((ext_vector_type(16))) __bf16 bf16x16;
typedef __attribute__((ext_vector_type(8)))  __bf16 bf16x8;
typedef __attribute__((ext_vector_type(4)))  __bf16 bf16x4;
typedef __attribute__((ext_vector_type(8)))  float  f32x8;
typedef __attribute__((ext_vector_type(4)))  float  f32x4;

// Fragment gather, CDNA5 ISA 7.12.2: element e of a lane's v16bf has
// K = 16*(e/8) + 8*(lane/16) + e%8 -> two contiguous 8-element (16B) runs.
// rowbase = &T[(lane%16 + tile_base) * TILE_LD + k0].
__device__ __forceinline__ bf16x16 load_frag(const __bf16* rowbase, int lane) {
  int h8 = ((lane >> 4) << 3);
  bf16x8 lo = *(const bf16x8*)(rowbase + h8);
  bf16x8 hi = *(const bf16x8*)(rowbase + 16 + h8);
  return __builtin_shufflevector(lo, hi, 0, 1, 2, 3, 4, 5, 6, 7,
                                 8, 9, 10, 11, 12, 13, 14, 15);
}

// 64x64x64 GEMM. A:[m][k], Bt:[n][k] (both K-contiguous, stride TILE_LD).
// Each wave owns rows [wave*16, wave*16+16): 4 f32 accumulators, 8 WMMAs.
__device__ __forceinline__ void gemm64(const __bf16* A, const __bf16* Bt,
                                       f32x8 acc[4], int wave, int lane) {
  const __bf16* arow = A + (wave * 16 + (lane & 15)) * TILE_LD;
  const __bf16* brow = Bt + (lane & 15) * TILE_LD;
#pragma unroll
  for (int kk = 0; kk < 64; kk += 32) {
    bf16x16 af = load_frag(arow + kk, lane);
#pragma unroll
    for (int nt = 0; nt < 4; ++nt) {
      bf16x16 bf = load_frag(brow + nt * 16 * TILE_LD + kk, lane);
      acc[nt] = __builtin_amdgcn_wmma_f32_16x16x32_bf16(
          false, af, false, bf, (short)0, acc[nt], false, false);
    }
  }
}

// ---- staging: 64x64 tiles global -> LDS bf16, vectorized ------------------

__device__ __forceinline__ void stage_f32(const float* src, size_t rs,
                                          __bf16* dst, int tid) {
  for (int idx = tid; idx < 1024; idx += 128) {
    int r = idx >> 4, c = (idx & 15) << 2;
    f32x4 v = *(const f32x4*)(src + (size_t)r * rs + c);
    bf16x4 b = {(__bf16)v.x, (__bf16)v.y, (__bf16)v.z, (__bf16)v.w};
    *(bf16x4*)(dst + r * TILE_LD + c) = b;
  }
}

__device__ __forceinline__ void stage_f32_T(const float* src, size_t rs,
                                            __bf16* dstT, int tid) {
  for (int idx = tid; idx < 1024; idx += 128) {
    int r = idx >> 4, c = (idx & 15) << 2;
    f32x4 v = *(const f32x4*)(src + (size_t)r * rs + c);
    dstT[(c + 0) * TILE_LD + r] = (__bf16)v.x;
    dstT[(c + 1) * TILE_LD + r] = (__bf16)v.y;
    dstT[(c + 2) * TILE_LD + r] = (__bf16)v.z;
    dstT[(c + 3) * TILE_LD + r] = (__bf16)v.w;
  }
}

__device__ __forceinline__ void stage_f32_both(const float* src, size_t rs,
                                               __bf16* dst, __bf16* dstT, int tid) {
  for (int idx = tid; idx < 1024; idx += 128) {
    int r = idx >> 4, c = (idx & 15) << 2;
    f32x4 v = *(const f32x4*)(src + (size_t)r * rs + c);
    bf16x4 b = {(__bf16)v.x, (__bf16)v.y, (__bf16)v.z, (__bf16)v.w};
    *(bf16x4*)(dst + r * TILE_LD + c) = b;
    dstT[(c + 0) * TILE_LD + r] = b.x;
    dstT[(c + 1) * TILE_LD + r] = b.y;
    dstT[(c + 2) * TILE_LD + r] = b.z;
    dstT[(c + 3) * TILE_LD + r] = b.w;
  }
}

__device__ __forceinline__ void stage_bf16(const __bf16* src, size_t rs,
                                           __bf16* dst, int tid) {
  for (int idx = tid; idx < 1024; idx += 128) {
    int r = idx >> 4, c = (idx & 15) << 2;
    *(bf16x4*)(dst + r * TILE_LD + c) = *(const bf16x4*)(src + (size_t)r * rs + c);
  }
}

__device__ __forceinline__ void stage_bf16_T(const __bf16* src, size_t rs,
                                             __bf16* dstT, int tid) {
  for (int idx = tid; idx < 1024; idx += 128) {
    int r = idx >> 4, c = (idx & 15) << 2;
    bf16x4 v = *(const bf16x4*)(src + (size_t)r * rs + c);
    dstT[(c + 0) * TILE_LD + r] = v.x;
    dstT[(c + 1) * TILE_LD + r] = v.y;
    dstT[(c + 2) * TILE_LD + r] = v.z;
    dstT[(c + 3) * TILE_LD + r] = v.w;
  }
}

// ---- accumulator writeback (C/D layout: row = r + 8*(lane/16), col = lane%16)

__device__ __forceinline__ void acc_to_lds(const f32x8 acc[4], float* S,
                                           float scale, const float* addcol,
                                           int wave, int lane) {
  int rbase = wave * 16 + ((lane >> 4) << 3);
  int c0 = lane & 15;
#pragma unroll
  for (int nt = 0; nt < 4; ++nt) {
    int col = nt * 16 + c0;
    float add = addcol ? addcol[col] : 0.0f;
#pragma unroll
    for (int r = 0; r < 8; ++r)
      S[(rbase + r) * S_LD + col] = acc[nt][r] * scale + add;
  }
}

__device__ __forceinline__ void acc_to_global_bf16(const f32x8 acc[4],
                                                   __bf16* dst, int stride,
                                                   int wave, int lane) {
  int rbase = wave * 16 + ((lane >> 4) << 3);
  int c0 = lane & 15;
#pragma unroll
  for (int nt = 0; nt < 4; ++nt)
#pragma unroll
    for (int r = 0; r < 8; ++r)
      dst[(size_t)(rbase + r) * stride + nt * 16 + c0] = (__bf16)acc[nt][r];
}

// ---------------------------------------------------------------------------

__global__ void k_fill_L(__bf16* L) {
  size_t i = (size_t)blockIdx.x * blockDim.x + threadIdx.x;
  L[i] = (__bf16)0.015625f;  // 1/m
}

// qbar[s][j][d] = sum_i L[s][i][j] * Q[i][s][d] / colsum_i(L).  Grid: bh*64+s.
__global__ void k_qbar(const float* __restrict__ Q, const __bf16* __restrict__ L,
                       __bf16* __restrict__ qbar) {
  __shared__ __align__(16) __bf16 sLT[64 * TILE_LD];  // [j][i]
  __shared__ __align__(16) __bf16 sQT[64 * TILE_LD];  // [d][i]
  __shared__ float winv[64];
  int bid = blockIdx.x, bh = bid >> 6, s = bid & 63;
  int tid = threadIdx.x, lane = tid & 31, wave = tid >> 5;

  stage_bf16_T(L + (size_t)bid * 4096, 64, sLT, tid);                  // L:[i][j]
  stage_f32_T(Q + (size_t)bh * 262144 + (size_t)s * 64, 4096, sQT, tid);
  __syncthreads();
  if (tid < 64) {  // colsum over i = contiguous row of sLT
    float wsum = 0.0f;
    for (int i = 0; i < 64; ++i) wsum += (float)sLT[tid * TILE_LD + i];
    winv[tid] = 1.0f / wsum;
  }
  __syncthreads();

  f32x8 acc[4] = {};
  gemm64(sLT, sQT, acc, wave, lane);  // A[m=j][k=i], Bt[n=d][k=i]

  __bf16* dst = qbar + (size_t)bid * 4096;  // [j][d]
  int rbase = wave * 16 + ((lane >> 4) << 3);
  int c0 = lane & 15;
#pragma unroll
  for (int nt = 0; nt < 4; ++nt)
#pragma unroll
    for (int r = 0; r < 8; ++r) {
      int j = rbase + r;
      dst[(size_t)j * 64 + nt * 16 + c0] = (__bf16)(acc[nt][r] * winv[j]);
    }
}

// Per (bh, j): S = scale*qbar_j*K_j^T -> softmax_t -> R, ent; kbar = R*K_j;
// final step also computes vbar = R*V_j (R never touches global memory).
__global__ void k_R_kbar(const float* __restrict__ Kg, const float* __restrict__ Vg,
                         const __bf16* __restrict__ qbar,
                         __bf16* __restrict__ kbar, __bf16* __restrict__ vbar,
                         float* __restrict__ ent, int final_step) {
  __shared__ __align__(16) __bf16 sK[64 * TILE_LD];   // [t][d] (Bt of GEMM1)
  __shared__ __align__(16) __bf16 sKT[64 * TILE_LD];  // [d][t] (Bt of GEMM2)
  __shared__ __align__(16) __bf16 sX[64 * TILE_LD];   // qbar [s][d]; later V^T [d][t]
  __shared__ __align__(16) __bf16 sR[64 * TILE_LD];   // [s][t]
  __shared__ float sS[64 * S_LD];
  int bid = blockIdx.x, bh = bid >> 6, j = bid & 63;
  int tid = threadIdx.x, lane = tid & 31, wave = tid >> 5;

  stage_f32_both(Kg + (size_t)bh * 262144 + (size_t)j * 4096, 64, sK, sKT, tid);
  stage_bf16(qbar + (size_t)bh * 262144 + (size_t)j * 64, 4096, sX, tid);
  __syncthreads();

  f32x8 acc[4] = {};
  gemm64(sX, sK, acc, wave, lane);                   // S[s,t] = qb.K^T
  acc_to_lds(acc, sS, 0.125f, nullptr, wave, lane);  // scale = 1/sqrt(64)
  __syncthreads();

  if (tid < 64) {  // softmax over t for row s = tid; entropy
    int s = tid;
    float mx = -1e30f;
    for (int t = 0; t < 64; ++t) mx = fmaxf(mx, sS[s * S_LD + t]);
    float sum = 0.0f, dot = 0.0f;
    for (int t = 0; t < 64; ++t) {
      float z = sS[s * S_LD + t];
      float e = __expf(z - mx);
      sum += e; dot += e * z;
      sS[s * S_LD + t] = e;
    }
    float inv = 1.0f / sum;
    for (int t = 0; t < 64; ++t) sR[s * TILE_LD + t] = (__bf16)(sS[s * S_LD + t] * inv);
    ent[(size_t)bid * 64 + s] = __logf(sum) + mx - dot * inv;
  }
  __syncthreads();

  f32x8 acc2[4] = {};
  gemm64(sR, sKT, acc2, wave, lane);  // kbar[s,d] = R*K
  acc_to_global_bf16(acc2, kbar + (size_t)bid * 4096, 64, wave, lane);

  if (final_step) {
    __syncthreads();
    stage_f32_T(Vg + (size_t)bh * 262144 + (size_t)j * 4096, 64, sX, tid);  // V^T [d][t]
    __syncthreads();
    f32x8 acc3[4] = {};
    gemm64(sR, sX, acc3, wave, lane);  // vbar[s,d] = R*V
    acc_to_global_bf16(acc3, vbar + (size_t)bid * 4096, 64, wave, lane);
  }
}

// Per (bh, s): scores[i,j] = scale*Q[i,:].kbar[j,s,:] + ent[j,s]; softmax_j -> L
__global__ void k_L(const float* __restrict__ Qg, const __bf16* __restrict__ kbar,
                    const float* __restrict__ ent, __bf16* __restrict__ L) {
  __shared__ __align__(16) __bf16 sQ[64 * TILE_LD];   // [i][d]
  __shared__ __align__(16) __bf16 skb[64 * TILE_LD];  // [j][d]
  __shared__ float sS[64 * S_LD];
  __shared__ float sE[64];
  int bid = blockIdx.x, bh = bid >> 6, s = bid & 63;
  int tid = threadIdx.x, lane = tid & 31, wave = tid >> 5;

  stage_f32(Qg + (size_t)bh * 262144 + (size_t)s * 64, 4096, sQ, tid);
  stage_bf16(kbar + (size_t)bh * 262144 + (size_t)s * 64, 4096, skb, tid);
  if (tid < 64) sE[tid] = ent[((size_t)bh * 64 + tid) * 64 + s];
  __syncthreads();

  f32x8 acc[4] = {};
  gemm64(sQ, skb, acc, wave, lane);             // A[i][d], Bt[j][d]
  acc_to_lds(acc, sS, 0.125f, sE, wave, lane);  // + ent[j]
  __syncthreads();

  if (tid < 64) {  // softmax over j for row i = tid
    int i = tid;
    float mx = -1e30f;
    for (int jj = 0; jj < 64; ++jj) mx = fmaxf(mx, sS[i * S_LD + jj]);
    float sum = 0.0f;
    for (int jj = 0; jj < 64; ++jj) {
      float e = __expf(sS[i * S_LD + jj] - mx);
      sS[i * S_LD + jj] = e;
      sum += e;
    }
    float inv = 1.0f / sum;
    __bf16* dst = L + (size_t)bid * 4096 + (size_t)i * 64;
    for (int jj = 0; jj < 64; ++jj) dst[jj] = (__bf16)(sS[i * S_LD + jj] * inv);
  }
}

// Per (bh, s): out[i, s, d] = sum_j L[s][i][j] * vbar[j][s][d]
__global__ void k_out(const __bf16* __restrict__ L, const __bf16* __restrict__ vbar,
                      float* __restrict__ out) {
  __shared__ __align__(16) __bf16 sL[64 * TILE_LD];   // [i][j]
  __shared__ __align__(16) __bf16 sVT[64 * TILE_LD];  // [d][j]
  int bid = blockIdx.x, bh = bid >> 6, s = bid & 63;
  int tid = threadIdx.x, lane = tid & 31, wave = tid >> 5;

  stage_bf16(L + (size_t)bid * 4096, 64, sL, tid);
  stage_bf16_T(vbar + (size_t)bh * 262144 + (size_t)s * 64, 4096, sVT, tid);
  __syncthreads();

  f32x8 acc[4] = {};
  gemm64(sL, sVT, acc, wave, lane);  // A[i][j], Bt[d][j]

  float* dst = out + (size_t)bh * 262144 + (size_t)s * 64;  // + i*4096 + d
  int rbase = wave * 16 + ((lane >> 4) << 3);
  int c0 = lane & 15;
#pragma unroll
  for (int nt = 0; nt < 4; ++nt)
#pragma unroll
    for (int r = 0; r < 8; ++r)
      dst[(size_t)(rbase + r) * 4096 + nt * 16 + c0] = acc[nt][r];
}

// ---------------------------------------------------------------------------

extern "C" void kernel_launch(void* const* d_in, const int* in_sizes, int n_in,
                              void* d_out, int out_size, void* d_ws, size_t ws_size,
                              hipStream_t stream) {
  (void)in_sizes; (void)n_in; (void)out_size; (void)ws_size;
  const float* Q = (const float*)d_in[0];
  const float* K = (const float*)d_in[1];
  const float* V = (const float*)d_in[2];
  float* out = (float*)d_out;

  char* ws = (char*)d_ws;
  const size_t big = 33554432;  // 64 BH * 64*64*64 bf16 * 2B
  __bf16* Lbuf = (__bf16*)(ws);
  __bf16* qbar = (__bf16*)(ws + big);
  __bf16* kbar = (__bf16*)(ws + 2 * big);
  __bf16* vbar = (__bf16*)(ws + 3 * big);
  float* ent = (float*)(ws + 4 * big);  // +1MB; ~135MB total workspace

  k_fill_L<<<65536, 256, 0, stream>>>(Lbuf);
  for (int step = 0; step < 3; ++step) {
    int fin = (step == 2);
    k_qbar<<<4096, 128, 0, stream>>>(Q, Lbuf, qbar);
    k_R_kbar<<<4096, 128, 0, stream>>>(K, V, qbar, kbar, vbar, ent, fin);
    k_L<<<4096, 128, 0, stream>>>(Q, kbar, ent, Lbuf);
  }
  k_out<<<4096, 128, 0, stream>>>(Lbuf, vbar, out);
}
